// MtanClassif_87565793231274
// MI455X (gfx1250) — matrix operand order repaired and checked by reading, not compiled, not run on hardware
//
#include <hip/hip_runtime.h>
#include <hip/hip_bf16.h>

typedef __attribute__((ext_vector_type(16))) _Float16 v16h;
typedef __attribute__((ext_vector_type(8)))  _Float16 v8h;
typedef __attribute__((ext_vector_type(8)))  float    v8f;

union V16 { v16h v; v8h h[2]; };

// ---------------------------------------------------------------------------
// Stage 1: time embedding -> f16 rows of 128  (j==0: linear, j>0: sin features)
// ---------------------------------------------------------------------------
__global__ void embed_kernel(const float* __restrict__ times, int rows,
                             const float* __restrict__ w_lin, const float* __restrict__ b_lin,
                             const float* __restrict__ w_per, const float* __restrict__ b_per,
                             _Float16* __restrict__ out) {
    int idx = blockIdx.x * blockDim.x + threadIdx.x;
    if (idx >= rows * 128) return;
    int row = idx >> 7, j = idx & 127;
    float t = times[row];
    float val = (j == 0) ? (t * w_lin[0] + b_lin[0])
                         : sinf(t * w_per[j - 1] + b_per[j - 1]);
    out[idx] = (_Float16)val;
}

// f32 [128][128] -> f16 transposed [n][k]  (so WMMA B-operand loads are
// contiguous 16-byte v8h loads per lane)
__global__ void f32_to_f16_T_kernel(const float* __restrict__ src,
                                    _Float16* __restrict__ dst) {
    int idx = blockIdx.x * blockDim.x + threadIdx.x;   // < 128*128
    int n = idx >> 7, k = idx & 127;
    dst[idx] = (_Float16)src[k * 128 + n];
}

// ---------------------------------------------------------------------------
// Stage 2: projection GEMM  out[R,128] = (A[R,128] @ W[128,128] + bias)*scale
// Wt is pre-transposed f16 [n][k]. One 16x16 tile per wave; 8 waves/block
// cover the 8 n-tiles of one row tile; 4 k-steps of 32.
// ---------------------------------------------------------------------------
__global__ void proj_wmma_kernel(const _Float16* __restrict__ A,
                                 const _Float16* __restrict__ Wt,
                                 const float* __restrict__ bias,
                                 float scale,
                                 _Float16* __restrict__ out) {
    int lane = threadIdx.x & 31;
    int nt   = threadIdx.x >> 5;       // 8 waves -> 8 n-tiles
    int rt   = blockIdx.x;             // row tile
    int m    = lane & 15;
    int hi   = lane >> 4;
    int kb8  = hi << 3;
    int kb16 = hi << 4;

    v8f acc = {};
    for (int ks = 0; ks < 4; ++ks) {
        int k0 = ks * 32;
        V16 a;
        const _Float16* ap = A + (rt * 16 + m) * 128 + k0 + kb8;
        a.h[0] = *(const v8h*)ap;
        a.h[1] = *(const v8h*)(ap + 16);
        V16 bm;
        const _Float16* wp = Wt + (nt * 16 + m) * 128 + k0 + kb8;
        bm.h[0] = *(const v8h*)wp;
        bm.h[1] = *(const v8h*)(wp + 16);
        acc = __builtin_amdgcn_wmma_f32_16x16x32_f16(false, a.v, false, bm.v,
                                                     (short)0, acc, false, false);
    }
#pragma unroll
    for (int r = 0; r < 8; ++r) {
        int row = rt * 16 + kb8 + r;
        int col = nt * 16 + m;
        out[row * 128 + col] = (_Float16)((acc[r] + bias[col]) * scale);
    }
}

// ---------------------------------------------------------------------------
// Stage 3: scores[bh, q, k] = Q_h[q,:16] . K_h[k,:16]   (1/sqrt(DK) folded
// into Q).  K-dim padded 16 -> 32 with zeros.  64 blocks (one per b,h),
// 8 waves: wave w -> q-tile w/2, k-tiles (w&1)*8 .. +7.
// ---------------------------------------------------------------------------
__global__ void scores_wmma_kernel(const _Float16* __restrict__ Qh,
                                   const _Float16* __restrict__ Kh,
                                   float* __restrict__ scores) {
    int bh = blockIdx.x; int b = bh >> 3; int h = bh & 7;
    int lane = threadIdx.x & 31;
    int wave = threadIdx.x >> 5;
    int qt  = wave >> 1;
    int kt0 = (wave & 1) * 8;
    int m = lane & 15, hi = lane >> 4, kb8 = hi << 3;
    v8h z8 = {};

    V16 a;                                        // A: Q tile 16x32 (K 16..31 = pad)
    a.h[0] = *(const v8h*)(Qh + (qt * 16 + m) * 128 + h * 16 + kb8);
    a.h[1] = z8;

    for (int kk = 0; kk < 8; ++kk) {
        int key0 = (kt0 + kk) * 16;
        V16 bm;                                   // B: K^T tile 32x16, K=d index
        const _Float16* kp = Kh + (b * 256 + key0 + m) * 128 + h * 16;
        bm.h[0] = *(const v8h*)(kp + kb8);
        bm.h[1] = z8;
        v8f acc = {};
        acc = __builtin_amdgcn_wmma_f32_16x16x32_f16(false, a.v, false, bm.v,
                                                     (short)0, acc, false, false);
#pragma unroll
        for (int r = 0; r < 8; ++r)
            scores[(bh * 64 + qt * 16 + kb8 + r) * 256 + key0 + m] = acc[r];
    }
}

// ---------------------------------------------------------------------------
// Stage 4a: per-row max + exp -> f16.  One wave per (b,h,q) row of 256.
// ---------------------------------------------------------------------------
__global__ void row_exp_kernel(const float* __restrict__ scores,
                               _Float16* __restrict__ eh) {
    int lane = threadIdx.x & 31;
    int wave = threadIdx.x >> 5;
    int row  = blockIdx.x * 4 + wave;     // 4096 rows
    const float* sp = scores + row * 256 + lane * 8;
    float v[8]; float mx = -3.4e38f;
#pragma unroll
    for (int e = 0; e < 8; ++e) { v[e] = sp[e]; mx = fmaxf(mx, v[e]); }
#pragma unroll
    for (int off = 16; off > 0; off >>= 1)
        mx = fmaxf(mx, __shfl_xor(mx, off, 32));
    _Float16* op = eh + row * 256 + lane * 8;
#pragma unroll
    for (int e = 0; e < 8; ++e) op[e] = (_Float16)__expf(v[e] - mx);
}

// ---------------------------------------------------------------------------
// Stage 4b: transposed combined value/mask matrix
//   w2t[b][c][k],  c<16: mask*x ; 16<=c<32: mask ; c>=32: mask (denominator)
// Indexed by output so stores are contiguous.
// ---------------------------------------------------------------------------
__global__ void w2_prep_kernel(const float* __restrict__ x,
                               const int* __restrict__ mask,
                               _Float16* __restrict__ w2t) {
    int idx = blockIdx.x * blockDim.x + threadIdx.x;   // < 8*64*256
    int k = idx & 255;
    int c = (idx >> 8) & 63;
    int b = idx >> 14;
    int mv = mask[(b * 256 + k) * 16 + (c & 15)];
    float w = (c < 16) ? (mv ? x[(b * 256 + k) * 16 + c] : 0.0f) : (float)mv;
    w2t[idx] = (_Float16)w;
}

// ---------------------------------------------------------------------------
// Stage 4c: attention GEMM per (b,h):  att[64,64] = e[64,256] @ W2_b[256,64]
// cols 0..31 = numerator, 32..63 = denominator.  w2t is transposed [b][c][k]
// so B loads are contiguous.  64 blocks, 4 waves; wave = q-tile; A reused
// across 4 n-tiles; 8 k-steps of 32.
// ---------------------------------------------------------------------------
__global__ void attn_wmma_kernel(const _Float16* __restrict__ eh,
                                 const _Float16* __restrict__ w2t,
                                 float* __restrict__ att) {
    int bh = blockIdx.x; int b = bh >> 3;
    int lane = threadIdx.x & 31;
    int qt   = threadIdx.x >> 5;       // 4 waves
    int m = lane & 15, hi = lane >> 4, kb8 = hi << 3, kb16 = hi << 4;

    v8f zf = {};
    v8f acc[4];
#pragma unroll
    for (int nt = 0; nt < 4; ++nt) acc[nt] = zf;

    for (int ks = 0; ks < 8; ++ks) {
        int k0 = ks * 32;
        V16 a;
        const _Float16* ap = eh + (bh * 64 + qt * 16 + m) * 256 + k0 + kb8;
        a.h[0] = *(const v8h*)ap;
        a.h[1] = *(const v8h*)(ap + 16);
#pragma unroll
        for (int nt = 0; nt < 4; ++nt) {
            V16 bm;
            const _Float16* wp = w2t + (b * 64 + nt * 16 + m) * 256 + k0 + kb8;
            bm.h[0] = *(const v8h*)wp;
            bm.h[1] = *(const v8h*)(wp + 16);
            acc[nt] = __builtin_amdgcn_wmma_f32_16x16x32_f16(false, a.v, false, bm.v,
                                                             (short)0, acc[nt], false, false);
        }
    }
#pragma unroll
    for (int nt = 0; nt < 4; ++nt)
#pragma unroll
        for (int r = 0; r < 8; ++r)
            att[(bh * 64 + qt * 16 + kb8 + r) * 64 + nt * 16 + m] = acc[nt][r];
}

// ---------------------------------------------------------------------------
// Stage 5: out[b,q,j] = bo[j] + sum_c (num/den)[b,h(c),q,c&31] * Wo[c,j]
// ---------------------------------------------------------------------------
__global__ void final_proj_kernel(const float* __restrict__ att,
                                  const float* __restrict__ Wo,
                                  const float* __restrict__ bo,
                                  float* __restrict__ out) {
    int idx = blockIdx.x * blockDim.x + threadIdx.x;
    if (idx >= 8 * 64 * 5) return;
    int j = idx % 5;
    int q = (idx / 5) & 63;
    int b = idx / (5 * 64);
    float acc = bo[j];
    for (int c = 0; c < 256; ++c) {
        int h = c >> 5, cc = c & 31;
        const float* ap = att + ((b * 8 + h) * 64 + q) * 64;
        float num = ap[cc];
        float den = ap[32 + cc];
        float r = (den > 0.0f) ? (num / den) : 0.0f;
        acc += r * Wo[c * 5 + j];
    }
    out[idx] = acc;
}

// ---------------------------------------------------------------------------
extern "C" void kernel_launch(void* const* d_in, const int* in_sizes, int n_in,
                              void* d_out, int out_size, void* d_ws, size_t ws_size,
                              hipStream_t stream) {
    const float* x     = (const float*)d_in[0];
    const float* ts    = (const float*)d_in[1];
    const int*   mask  = (const int*)  d_in[2];
    const float* qtt   = (const float*)d_in[3];
    const float* w_lin = (const float*)d_in[4];
    const float* b_lin = (const float*)d_in[5];
    const float* w_per = (const float*)d_in[6];
    const float* b_per = (const float*)d_in[7];
    const float* Wq    = (const float*)d_in[8];
    const float* bq    = (const float*)d_in[9];
    const float* Wk    = (const float*)d_in[10];
    const float* bk    = (const float*)d_in[11];
    const float* Wo    = (const float*)d_in[12];
    const float* bo    = (const float*)d_in[13];
    float* out = (float*)d_out;

    char* ws = (char*)d_ws;
    size_t off = 0;
    auto carve = [&](size_t bytes) {
        char* p = ws + off;
        off = (off + bytes + 255) & ~(size_t)255;
        return p;
    };
    _Float16* qemb  = (_Float16*)carve((size_t)64 * 128 * 2);
    _Float16* kemb  = (_Float16*)carve((size_t)2048 * 128 * 2);
    _Float16* wqt   = (_Float16*)carve((size_t)128 * 128 * 2);
    _Float16* wkt   = (_Float16*)carve((size_t)128 * 128 * 2);
    _Float16* qh    = (_Float16*)carve((size_t)64 * 128 * 2);
    _Float16* kh    = (_Float16*)carve((size_t)2048 * 128 * 2);
    float*    sc    = (float*)   carve((size_t)4096 * 256 * 4);
    _Float16* ehalf = (_Float16*)carve((size_t)4096 * 256 * 2);
    _Float16* w2t   = (_Float16*)carve((size_t)8 * 64 * 256 * 2);
    float*    att   = (float*)   carve((size_t)4096 * 64 * 4);
    (void)ws_size; (void)in_sizes; (void)n_in; (void)out_size;

    embed_kernel<<<(2048 * 128) / 256, 256, 0, stream>>>(ts, 2048, w_lin, b_lin, w_per, b_per, kemb);
    embed_kernel<<<(64 * 128) / 256, 256, 0, stream>>>(qtt, 64, w_lin, b_lin, w_per, b_per, qemb);
    f32_to_f16_T_kernel<<<(128 * 128) / 256, 256, 0, stream>>>(Wq, wqt);
    f32_to_f16_T_kernel<<<(128 * 128) / 256, 256, 0, stream>>>(Wk, wkt);
    proj_wmma_kernel<<<2048 / 16, 256, 0, stream>>>(kemb, wkt, bk, 1.0f, kh);
    proj_wmma_kernel<<<64 / 16, 256, 0, stream>>>(qemb, wqt, bq, 0.25f, qh);  // 1/sqrt(16)
    scores_wmma_kernel<<<64, 256, 0, stream>>>(qh, kh, sc);
    row_exp_kernel<<<1024, 128, 0, stream>>>(sc, ehalf);
    w2_prep_kernel<<<(8 * 64 * 256) / 256, 256, 0, stream>>>(x, mask, w2t);
    attn_wmma_kernel<<<64, 128, 0, stream>>>(ehalf, w2t, att);
    final_proj_kernel<<<10, 256, 0, stream>>>(att, Wo, bo, out);
}
